// NormAttention_50225347559957
// MI455X (gfx1250) — matrix operand order repaired and checked by reading, not compiled, not run on hardware
//
#include <hip/hip_runtime.h>
#include <hip/hip_bf16.h>

#define BATCH 2
#define SEQ   2048
#define DIM   2048
#define NH    16
#define NKV   4
#define HD    128
#define QKV_E ((NH + 2*NKV)*HD)   /* 3072 */
#define EPSV  1e-6f

typedef __bf16 bf16;
typedef __attribute__((ext_vector_type(16))) __bf16 v16bf;
typedef __attribute__((ext_vector_type(8)))  __bf16 v8bf;
typedef __attribute__((ext_vector_type(8)))  float  v8f;
typedef __attribute__((ext_vector_type(4)))  unsigned int v4u;
typedef __attribute__((ext_vector_type(8)))  int v8i;
typedef __attribute__((ext_vector_type(4)))  int v4i;

// ---- Tensor Data Mover availability (probe-verified builtins; arity differs per toolchain) ----
#if defined(__has_builtin)
#if __has_builtin(__builtin_amdgcn_tensor_load_to_lds) && __has_builtin(__builtin_amdgcn_s_wait_tensorcnt)
#define USE_TDM 1
#endif
#endif
#ifndef USE_TDM
#define USE_TDM 0
#endif

#if USE_TDM
#if __clang_major__ >= 23
#define TDM_LOAD(g0, g1) __builtin_amdgcn_tensor_load_to_lds((g0), (g1), (v4i)0, (v4i)0, (v8i)0, 0)
#else
#define TDM_LOAD(g0, g1) __builtin_amdgcn_tensor_load_to_lds((g0), (g1), (v4i)0, (v4i)0, 0)
#endif

// Build a 2-D tile descriptor (D#) per CDNA5 ISA §8.3/§8.4 and issue TENSOR_LOAD_TO_LDS.
//   src          : global address of tile start (bf16 elements, row-major)
//   lds_byte_off : LDS byte address of destination (tile stored densely, row-major)
//   rows         : tile_dim1 (# rows),  rowlen: tile_dim0 (elements per row, contiguous)
//   stride_elems : tensor_dim0_stride (row pitch in elements)
__device__ __forceinline__ void tdm_load_2d(const bf16* src, unsigned lds_byte_off,
                                            unsigned rows, unsigned rowlen,
                                            unsigned stride_elems, unsigned tensor_rows) {
    unsigned long long ga = (unsigned long long)src;
    v4u g0;
    g0[0] = 1u;                                            // count=1, user descriptor
    g0[1] = lds_byte_off;                                  // lds_addr  [63:32]
    g0[2] = (unsigned)(ga & 0xffffffffu);                  // global_addr[95:64]
    g0[3] = (unsigned)((ga >> 32) & 0x01ffffffu)           // global_addr[120:96]
          | (2u << 30);                                    // type=2 ("image")
    unsigned td0 = stride_elems;                           // tensor_dim0 (generous, no OOB clip)
    unsigned td1 = tensor_rows;                            // tensor_dim1
    unsigned long long st0 = stride_elems;                 // tensor_dim0_stride
    v8i g1;
    g1[0] = (int)(1u << 16);                               // data_size=1 (2B); mask/pad/iter=0
    g1[1] = (int)((td0 & 0xffffu) << 16);                  // [47:32]=0, [63:48]=td0.lo16
    g1[2] = (int)((td0 >> 16) | ((td1 & 0xffffu) << 16));  // [79:64]=td0.hi16, [95:80]=td1.lo16
    g1[3] = (int)((td1 >> 16) | ((rowlen & 0xffffu) << 16)); // [111:96]=td1.hi16, [127:112]=tile_dim0
    g1[4] = (int)(rows & 0xffffu);                         // [143:128]=tile_dim1, [159:144]=tile_dim2=0
    g1[5] = (int)(st0 & 0xffffffffu);                      // [191:160]=stride.lo32
    g1[6] = (int)((st0 >> 32) & 0xffffu);                  // [207:192]=stride.hi16, dim1_stride=0
    g1[7] = 0;
    TDM_LOAD(g0, g1);
}

__device__ __forceinline__ unsigned lds_off32(const void* p) {
    // LDS objects live at {SHARED_BASE, offset}; low 32 bits are the LDS byte offset.
    return (unsigned)(unsigned long long)p;
}
#endif // USE_TDM

// ---------------- RMSNorm (fp32 in, bf16 out) ----------------
__global__ void rmsnorm_kernel(const float* __restrict__ x,
                               const float* __restrict__ w,
                               bf16* __restrict__ xn) {
    __shared__ float red[256];
    int row = blockIdx.x;                       // b*SEQ + s
    const float* xr = x + (size_t)row * DIM;
    float ss = 0.f;
    for (int i = threadIdx.x; i < DIM; i += 256) { float v = xr[i]; ss += v * v; }
    red[threadIdx.x] = ss; __syncthreads();
    for (int s = 128; s > 0; s >>= 1) {
        if (threadIdx.x < s) red[threadIdx.x] += red[threadIdx.x + s];
        __syncthreads();
    }
    float inv = rsqrtf(red[0] / (float)DIM + EPSV);
    bf16* o = xn + (size_t)row * DIM;
    for (int i = threadIdx.x; i < DIM; i += 256) o[i] = (bf16)(xr[i] * inv * w[i]);
}

// ---------------- fp32 -> bf16 cast ----------------
__global__ void cast_bf16_kernel(const float* __restrict__ s, bf16* __restrict__ d, int n) {
    int i = blockIdx.x * blockDim.x + threadIdx.x;
    int stride = gridDim.x * blockDim.x;
    for (; i < n; i += stride) d[i] = (bf16)s[i];
}

// ---------------- WMMA bf16 GEMM: C[M,N] = A[M,K] * Bt[N,K]^T ----------------
// TDM path: block = 8 waves (2x4), block tile 64x256, wave tile 32x64.
// Per 32-wide k-step the TDM DMAs A(64x32) and B(256x32) panels into LDS
// (double-buffered, TENSORcnt-pipelined); waves read fragments via ds_load.
// Fallback: per-wave direct global loads (register-blocked) + prefetch.
#if USE_TDM
template <typename OUT>
__global__ void __launch_bounds__(256)
gemm_bf16_wmma(const bf16* __restrict__ A, const bf16* __restrict__ Bt,
               OUT* __restrict__ C, int M, int N, int K) {
    __shared__ bf16 ldsA[2][64][32];            //  8 KiB
    __shared__ bf16 ldsB[2][256][32];           // 32 KiB
    int ntn = N >> 8;                           // N / 256
    int bm = blockIdx.x / ntn, bn = blockIdx.x % ntn;
    int wv = threadIdx.x >> 5, lane = threadIdx.x & 31;
    int wm = wv >> 2, wn = wv & 3;              // wave position: 2 x 4
    int half = lane >> 4, l15 = lane & 15;
    int NK = K >> 5;

    const bf16* Ab = A  + (size_t)(bm * 64)  * K;
    const bf16* Bb = Bt + (size_t)(bn * 256) * K;

    if (wv == 0)  {                             // stage panels for k-step 0
        tdm_load_2d(Ab,      lds_off32(&ldsA[0][0][0]), 64,  32, (unsigned)K, (unsigned)M);
        tdm_load_2d(Bb,      lds_off32(&ldsB[0][0][0]), 256, 32, (unsigned)K, (unsigned)N);
    }

    v8f acc[2][4] = {};
    for (int ks = 0; ks < NK; ks++) {
        int buf = ks & 1;
        if (wv == 0) __builtin_amdgcn_s_wait_tensorcnt((unsigned short)0); // panels(ks) landed
        __syncthreads();                        // all waves: panels ready; prior buffer reads done
        if (wv == 0 && ks + 1 < NK) {           // DMA next panels, overlapped with compute
            int k1 = (ks + 1) * 32;
            tdm_load_2d(Ab + k1, lds_off32(&ldsA[buf ^ 1][0][0]), 64,  32, (unsigned)K, (unsigned)M);
            tdm_load_2d(Bb + k1, lds_off32(&ldsB[buf ^ 1][0][0]), 256, 32, (unsigned)K, (unsigned)N);
        }
        // A fragment: lane holds M=l15, K contiguous 0..15 (half0) / 16..31 (half1)
        v16bf a[2];
#pragma unroll
        for (int im = 0; im < 2; im++) {
            const bf16* arow = &ldsA[buf][wm * 32 + im * 16 + l15][0] + (half ? 16 : 0);
            v8bf lo = *(const v8bf*)arow;
            v8bf hi = *(const v8bf*)(arow + 8);
#pragma unroll
            for (int i = 0; i < 8; i++) { a[im][i] = lo[i]; a[im][8 + i] = hi[i]; }
        }
        // B fragment: lane holds N=l15, K contiguous 0..15 (half0) / 16..31 (half1)
        v16bf b[4];
#pragma unroll
        for (int j = 0; j < 4; j++)
            b[j] = *(const v16bf*)(&ldsB[buf][wn * 64 + j * 16 + l15][0] + (half ? 16 : 0));
#pragma unroll
        for (int im = 0; im < 2; im++)
#pragma unroll
            for (int j = 0; j < 4; j++)
                acc[im][j] = __builtin_amdgcn_wmma_f32_16x16x32_bf16(
                    false, a[im], false, b[j], (short)0, acc[im][j], false, false);
    }
    // C layout: lane holds N=l15, rows M = half*8 + i
#pragma unroll
    for (int im = 0; im < 2; im++) {
        int mb = bm * 64 + wm * 32 + im * 16 + (half ? 8 : 0);
#pragma unroll
        for (int j = 0; j < 4; j++) {
            OUT* cr = C + (size_t)mb * N + bn * 256 + wn * 64 + j * 16 + l15;
#pragma unroll
            for (int i = 0; i < 8; i++) cr[(size_t)i * N] = (OUT)acc[im][j][i];
        }
    }
}
#else  // !USE_TDM — direct-load fallback (register-blocked)
template <typename OUT>
__global__ void __launch_bounds__(256)
gemm_bf16_wmma(const bf16* __restrict__ A, const bf16* __restrict__ Bt,
               OUT* __restrict__ C, int M, int N, int K) {
    int wave = (int)((blockIdx.x * blockDim.x + threadIdx.x) >> 5);
    int lane = threadIdx.x & 31;
    int ntn = N >> 6;
    // map 8 waves of a block onto the same 64x256 region as the TDM path
    int blk = wave >> 3, wv = wave & 7;
    int nbn = N >> 8;
    int tm = (blk / nbn) * 2 + (wv >> 2);
    int tn = (blk % nbn) * 4 + (wv & 3);
    if (tm >= (M >> 5) || tn >= ntn) return;
    int half = lane >> 4, l15 = lane & 15;
    const bf16* ar[2];
    ar[0] = A + (size_t)(tm * 32 + l15) * K + (half ? 16 : 0);
    ar[1] = ar[0] + (size_t)16 * K;
    const bf16* br[4];
#pragma unroll
    for (int j = 0; j < 4; j++)
        br[j] = Bt + (size_t)(tn * 64 + j * 16 + l15) * K + (half ? 16 : 0);
    v8f acc[2][4] = {};
    for (int k0 = 0; k0 < K; k0 += 32) {
        if (k0 + 512 < K) {
#pragma unroll
            for (int im = 0; im < 2; im++) __builtin_prefetch(ar[im] + k0 + 512, 0, 1);
#pragma unroll
            for (int j = 0; j < 4; j++)  __builtin_prefetch(br[j] + k0 + 512, 0, 1);
        }
        v16bf a[2];
#pragma unroll
        for (int im = 0; im < 2; im++) {
            v8bf lo = *(const v8bf*)(ar[im] + k0);
            v8bf hi = *(const v8bf*)(ar[im] + k0 + 8);
#pragma unroll
            for (int i = 0; i < 8; i++) { a[im][i] = lo[i]; a[im][8 + i] = hi[i]; }
        }
        v16bf b[4];
#pragma unroll
        for (int j = 0; j < 4; j++) b[j] = *(const v16bf*)(br[j] + k0);
#pragma unroll
        for (int im = 0; im < 2; im++)
#pragma unroll
            for (int j = 0; j < 4; j++)
                acc[im][j] = __builtin_amdgcn_wmma_f32_16x16x32_bf16(
                    false, a[im], false, b[j], (short)0, acc[im][j], false, false);
    }
#pragma unroll
    for (int im = 0; im < 2; im++) {
        int mb = tm * 32 + im * 16 + (half ? 8 : 0);
#pragma unroll
        for (int j = 0; j < 4; j++) {
            OUT* cr = C + (size_t)mb * N + tn * 64 + j * 16 + l15;
#pragma unroll
            for (int i = 0; i < 8; i++) cr[(size_t)i * N] = (OUT)acc[im][j][i];
        }
    }
}
#endif // USE_TDM

// ---------------- RoPE + repack: qkv[b,s,3072] -> Qh[b,h,s,d], Kh[b,g,s,d], Vt[b,g,d,s] ----------------
__global__ void rope_pack_kernel(const bf16* __restrict__ qkv, const float* __restrict__ freqs,
                                 bf16* __restrict__ Qh, bf16* __restrict__ Kh,
                                 bf16* __restrict__ Vt) {
    int bs = blockIdx.x;
    int b = bs / SEQ, s = bs % SEQ;
    const bf16* row = qkv + (size_t)bs * QKV_E;
    const float* fr = freqs + (size_t)s * HD;   // [64][2] cos/sin
    int t = threadIdx.x;
#pragma unroll
    for (int it = 0; it < 4; it++) {
        int p = t + it * 256;                   // 0..1023
        int h = p >> 6, j = p & 63;
        float xr = (float)row[h * HD + 2 * j];
        float xi = (float)row[h * HD + 2 * j + 1];
        float c = fr[2 * j], sn = fr[2 * j + 1];
        bf16* q = Qh + ((size_t)(b * NH + h) * SEQ + s) * HD + 2 * j;
        q[0] = (bf16)(xr * c - xi * sn);
        q[1] = (bf16)(xr * sn + xi * c);
    }
    {
        int g = t >> 6, j = t & 63;
        float xr = (float)row[NH * HD + g * HD + 2 * j];
        float xi = (float)row[NH * HD + g * HD + 2 * j + 1];
        float c = fr[2 * j], sn = fr[2 * j + 1];
        bf16* k = Kh + ((size_t)(b * NKV + g) * SEQ + s) * HD + 2 * j;
        k[0] = (bf16)(xr * c - xi * sn);
        k[1] = (bf16)(xr * sn + xi * c);
    }
#pragma unroll
    for (int it = 0; it < 2; it++) {
        int e = t + it * 256;                   // 0..511
        int g = e >> 7, d = e & 127;
        Vt[((size_t)(b * NKV + g) * HD + d) * SEQ + s] = row[(NH + NKV) * HD + e];
    }
}

// ---------------- Flash attention: one wave per (b, head, 16-query tile) ----------------
__global__ void __launch_bounds__(256)
attn_kernel(const bf16* __restrict__ Qh, const bf16* __restrict__ Kh,
            const bf16* __restrict__ Vt, bf16* __restrict__ Ao) {
    __shared__ bf16 plds[8][16][32];            // per-wave P staging (C-layout -> A-layout)
    int wv = threadIdx.x >> 5, lane = threadIdx.x & 31;
    int wave = blockIdx.x * 8 + wv;
    int qtile = wave & 127;                     // SEQ/16 = 128
    int h = (wave >> 7) & 15;
    int b = wave >> 11;
    int g = h >> 2;                             // n_rep = 4
    int q0 = qtile * 16;
    int half = lane >> 4, l15 = lane & 15;

    const bf16* qbase = Qh + ((size_t)(b * NH + h) * SEQ + q0 + l15) * HD + (half ? 16 : 0);
    v16bf aq[4];
#pragma unroll
    for (int dk = 0; dk < 4; dk++) {
        v8bf lo = *(const v8bf*)(qbase + dk * 32);
        v8bf hi = *(const v8bf*)(qbase + dk * 32 + 8);
#pragma unroll
        for (int i = 0; i < 8; i++) { aq[dk][i] = lo[i]; aq[dk][8 + i] = hi[i]; }
    }
    const bf16* kbase = Kh + (size_t)(b * NKV + g) * SEQ * HD;
    const bf16* vbase = Vt + (size_t)(b * NKV + g) * HD * SEQ;

    v8f o[8] = {};
    float mrow[8], lrow[8];
#pragma unroll
    for (int i = 0; i < 8; i++) { mrow[i] = -1e30f; lrow[i] = 0.f; }
    const float scale = 0.08838834764831845f;   // 1/sqrt(128)

    int kvend = q0 + 16;
    for (int kv0 = 0; kv0 < kvend; kv0 += 32) {
        v8f sfrag[2];
#pragma unroll
        for (int t = 0; t < 2; t++) {
            int kvi = kv0 + t * 16 + l15;
            int kvc = kvi <= q0 + 15 ? kvi : q0 + 15;
            const bf16* kr = kbase + (size_t)kvc * HD + (half ? 16 : 0);
            v8f s = {};
#pragma unroll
            for (int dk = 0; dk < 4; dk++) {
                v16bf bk = *(const v16bf*)(kr + dk * 32);
                s = __builtin_amdgcn_wmma_f32_16x16x32_bf16(false, aq[dk], false, bk,
                                                            (short)0, s, false, false);
            }
            sfrag[t] = s;
        }
        float bmax[8];
#pragma unroll
        for (int i = 0; i < 8; i++) {
            int m = half * 8 + i;
#pragma unroll
            for (int t = 0; t < 2; t++) {
                int col = kv0 + t * 16 + l15;
                float v = sfrag[t][i] * scale;
                sfrag[t][i] = (col <= q0 + m) ? v : -1e30f;
            }
            float mx = fmaxf(sfrag[0][i], sfrag[1][i]);
            mx = fmaxf(mx, __shfl_xor(mx, 1, 32));
            mx = fmaxf(mx, __shfl_xor(mx, 2, 32));
            mx = fmaxf(mx, __shfl_xor(mx, 4, 32));
            mx = fmaxf(mx, __shfl_xor(mx, 8, 32));
            bmax[i] = mx;
        }
        float alpha[8];
#pragma unroll
        for (int i = 0; i < 8; i++) {
            float mnew = fmaxf(mrow[i], bmax[i]);
            alpha[i] = __expf(mrow[i] - mnew);
            mrow[i] = mnew;
        }
#pragma unroll
        for (int i = 0; i < 8; i++) {
            float p0 = __expf(sfrag[0][i] - mrow[i]);
            float p1 = __expf(sfrag[1][i] - mrow[i]);
            sfrag[0][i] = p0; sfrag[1][i] = p1;
            float rs = p0 + p1;
            rs += __shfl_xor(rs, 1, 32);
            rs += __shfl_xor(rs, 2, 32);
            rs += __shfl_xor(rs, 4, 32);
            rs += __shfl_xor(rs, 8, 32);
            lrow[i] = lrow[i] * alpha[i] + rs;
        }
#pragma unroll
        for (int j = 0; j < 8; j++)
#pragma unroll
            for (int i = 0; i < 8; i++) o[j][i] *= alpha[i];

        int mbase = half * 8;
#pragma unroll
        for (int i = 0; i < 8; i++) {
            plds[wv][mbase + i][l15]      = (bf16)sfrag[0][i];
            plds[wv][mbase + i][16 + l15] = (bf16)sfrag[1][i];
        }
        v16bf ap;
        int ko = half * 16;
#pragma unroll
        for (int i = 0; i < 8; i++) {
            ap[i]     = plds[wv][l15][ko + i];
            ap[8 + i] = plds[wv][l15][ko + 8 + i];
        }
#pragma unroll
        for (int j = 0; j < 8; j++) {
            const bf16* vr = vbase + (size_t)(j * 16 + l15) * SEQ + kv0 + (half ? 16 : 0);
            v16bf bv = *(const v16bf*)vr;
            o[j] = __builtin_amdgcn_wmma_f32_16x16x32_bf16(false, ap, false, bv,
                                                           (short)0, o[j], false, false);
        }
    }
#pragma unroll
    for (int i = 0; i < 8; i++) {
        float inv = 1.f / lrow[i];
        int m = half * 8 + i;
        bf16* orow = Ao + ((size_t)b * SEQ + q0 + m) * (size_t)(NH * HD) + h * HD + l15;
#pragma unroll
        for (int j = 0; j < 8; j++) orow[j * 16] = (bf16)(o[j][i] * inv);
    }
}

// ---------------- host-side launch ----------------
extern "C" void kernel_launch(void* const* d_in, const int* in_sizes, int n_in,
                              void* d_out, int out_size, void* d_ws, size_t ws_size,
                              hipStream_t stream) {
    const float* x      = (const float*)d_in[0];
    const float* freqs  = (const float*)d_in[1];
    const float* norm_w = (const float*)d_in[2];
    const float* wqkv   = (const float*)d_in[3];
    const float* wo     = (const float*)d_in[4];
    float* out = (float*)d_out;

    char* ws = (char*)d_ws;
    size_t off = 0;
    auto alloc = [&](size_t bytes) -> char* {
        char* p = ws + off;
        off += (bytes + 255) & ~(size_t)255;
        return p;
    };
    bf16* xn     = (bf16*)alloc((size_t)BATCH * SEQ * DIM * 2);      // 16 MiB
    bf16* wqkv_b = (bf16*)alloc((size_t)QKV_E * DIM * 2);            // 12 MiB
    bf16* wo_b   = (bf16*)alloc((size_t)DIM * NH * HD * 2);          //  8 MiB
    bf16* qkv    = (bf16*)alloc((size_t)BATCH * SEQ * QKV_E * 2);    // 24 MiB
    bf16* Qh     = (bf16*)alloc((size_t)BATCH * NH * SEQ * HD * 2);  // 16 MiB
    bf16* Kh     = (bf16*)alloc((size_t)BATCH * NKV * SEQ * HD * 2); //  4 MiB
    bf16* Vt     = (bf16*)alloc((size_t)BATCH * NKV * HD * SEQ * 2); //  4 MiB
    bf16* Ao     = xn;   // xn is dead after the QKV GEMM; reuse for attention output

    rmsnorm_kernel<<<BATCH * SEQ, 256, 0, stream>>>(x, norm_w, xn);
    cast_bf16_kernel<<<2048, 256, 0, stream>>>(wqkv, wqkv_b, QKV_E * DIM);
    cast_bf16_kernel<<<2048, 256, 0, stream>>>(wo, wo_b, DIM * NH * HD);

    {   // QKV projection: [4096,3072] = xn[4096,2048] * wqkv^T
        int M = BATCH * SEQ, N = QKV_E, K = DIM;
        int blocks = (M / 64) * (N / 256);      // 768 blocks x 8 waves
        gemm_bf16_wmma<bf16><<<blocks, 256, 0, stream>>>(xn, wqkv_b, qkv, M, N, K);
    }
    rope_pack_kernel<<<BATCH * SEQ, 256, 0, stream>>>(qkv, freqs, Qh, Kh, Vt);
    attn_kernel<<<(BATCH * NH * (SEQ / 16)) / 8, 256, 0, stream>>>(Qh, Kh, Vt, Ao);
    {   // Output projection: [4096,2048] = Ao[4096,2048] * wo^T  (fp32 out)
        int M = BATCH * SEQ, N = DIM, K = NH * HD;
        int blocks = (M / 64) * (N / 256);      // 512 blocks x 8 waves
        gemm_bf16_wmma<float><<<blocks, 256, 0, stream>>>(Ao, wo_b, out, M, N, K);
    }
}
